// ExpertChoiceTokenNoisyTopkRouter_9517647528391
// MI455X (gfx1250) — matrix operand order repaired and checked by reading, not compiled, not run on hardware
//
#include <hip/hip_runtime.h>
#include <hip/hip_bf16.h>
#include <math.h>

// ---------------------------------------------------------------------------
// ExpertChoice noisy top-k router for MI455X (gfx1250, wave32).
// Phase 0: pack W_route/W_noise -> bf16 WMMA B-fragment buffer in ws (512 KB).
// Phase 1: fused dual GEMM via v_wmma_f32_16x16x32_bf16, B staged with
//          global_load_async_to_lds_b128 (ASYNCcnt) + double-buffered LDS.
// Phase 2: per-expert exact top-512 via 4-pass radix select + masked softmax.
// ---------------------------------------------------------------------------

typedef __attribute__((ext_vector_type(16))) __bf16 v16bf;
typedef __attribute__((ext_vector_type(8)))  float  v8f;
typedef __attribute__((ext_vector_type(4)))  float  v4f;

#define N_TOK 16384
#define D_EMB 2048
#define N_EXP 64
#define TOPK  512
#define KSTEPS (D_EMB / 32)          // 64
#define STEP_BYTES 8192              // per K-step: 4KB Wr frags + 4KB Wn frags

// Async global->LDS copy, 16B per lane (GLOBAL_LOAD_ASYNC_TO_LDS_B128, GV mode).
// The LDS destination is passed BOTH as a pointer operand (so the shared array
// escapes into the asm and the "memory" clobber aliases it -- otherwise LLVM
// proves "no stores to LDS" and folds the later ds_loads to undef) and as the
// raw LDS byte offset, recovered from the flat address low 32 bits
// (ISA 10.2: in the LDS aperture the LDS address is the low 32 address bits).
__device__ __forceinline__ void async_ld_b128(void* lds_ptr, const void* gptr) {
    unsigned lds_off = (unsigned)(uintptr_t)lds_ptr;
    asm volatile("global_load_async_to_lds_b128 %0, %1, off"
                 :: "v"(lds_off), "v"(gptr), "v"(lds_ptr) : "memory");
}
__device__ __forceinline__ void wait_async0() {
    asm volatile("s_wait_asynccnt 0x0" ::: "memory");
}

// ---------------------------------------------------------------------------
// Phase 0: pack W (fp32 [D,E]) into bf16 B-fragment layout.
// Per K-step s: 2048 bf16 for Wr then 2048 bf16 for Wn.
// Fragment position of (k,col): tile t=col>>4, lane=(col&15)+16*(k>=16),
// element i=k&15  (k order used here, same as A: v16bf element i == K = 16*laneHalf + i).
// ---------------------------------------------------------------------------
__global__ __launch_bounds__(256) void pack_w_kernel(
    const float* __restrict__ Wr, const float* __restrict__ Wn,
    __bf16* __restrict__ frag)
{
    int gid = blockIdx.x * 256 + threadIdx.x;   // 0 .. 131071
    int s   = gid >> 11;                        // K-step
    int r   = gid & 2047;
    int k   = r >> 6;                           // 0..31
    int col = r & 63;                           // 0..63
    int t     = col >> 4;
    int dlane = (col & 15) + ((k & 16) ? 16 : 0);
    int i     = k & 15;
    size_t off = (size_t)s * 4096 + (size_t)(t * 512 + dlane * 16 + i);
    size_t g   = (size_t)(s * 32 + k) * N_EXP + col;
    frag[off]        = (__bf16)Wr[g];
    frag[off + 2048] = (__bf16)Wn[g];
}

// ---------------------------------------------------------------------------
// Phase 1: block = 128 tokens x 64 experts, 256 threads (8 waves).
// Wave w owns token tile [tokenBase + 16w, +16); loops 4 expert tiles x
// {route, noise}. B fragments arrive in LDS via async DMA, double buffered.
// ---------------------------------------------------------------------------
__global__ __launch_bounds__(256) void router_gemm_kernel(
    const float* __restrict__ x,        // [N, D]
    const __bf16* __restrict__ frag,    // packed W fragments
    const float* __restrict__ br,       // [E]
    const float* __restrict__ bn,       // [E]
    const float* __restrict__ noise,    // [E, N]
    float* __restrict__ noisy)          // [E, N]
{
    __shared__ __align__(16) char ldsBuf[2][STEP_BYTES];

    const int tid   = threadIdx.x;
    const int wave  = tid >> 5;          // 0..7 : token tile
    const int lane  = tid & 31;
    const int cl    = lane & 15;
    const int hi    = lane >> 4;
    const int tokenBase = blockIdx.x * 128;
    const int aRow  = tokenBase + wave * 16 + cl;
    const int hbase = hi ? 16 : 0;       // A-fragment K offset for lane half

    v8f accR[4] = {}, accN[4] = {};

    const float* xRow = x + (size_t)aRow * D_EMB;
    const char*  fb   = (const char*)frag;

    // prologue: stage step 0 into buffer 0 (2 x b128 per thread = 8KB total)
    {
        const char* src = fb + (size_t)tid * 32;
        char*       dst = &ldsBuf[0][tid * 32];
        async_ld_b128(dst,      src);
        async_ld_b128(dst + 16, src + 16);
    }

    for (int s = 0; s < KSTEPS; ++s) {
        wait_async0();            // own async copies into buf[s&1] done
        __syncthreads();          // all waves' copies done -> buf[s&1] valid

        if (s + 1 < KSTEPS) {     // issue next step's DMA into the other buffer
            const char* src = fb + (size_t)(s + 1) * STEP_BYTES + tid * 32;
            char*       dst = &ldsBuf[(s + 1) & 1][tid * 32];
            async_ld_b128(dst,      src);
            async_ld_b128(dst + 16, src + 16);
        }

        // ---- A fragment: row aRow, K = kk + {hbase..hbase+15} ----
        const int kk = s * 32;
        v4f a0 = __builtin_nontemporal_load((const v4f*)(xRow + kk + hbase));
        v4f a1 = __builtin_nontemporal_load((const v4f*)(xRow + kk + hbase + 4));
        v4f a2 = __builtin_nontemporal_load((const v4f*)(xRow + kk + hbase + 8));
        v4f a3 = __builtin_nontemporal_load((const v4f*)(xRow + kk + hbase + 12));
        if (kk + 32 < D_EMB)
            __builtin_prefetch(xRow + kk + 32, 0, 1);   // global_prefetch_b8

        v16bf afrag;
        #pragma unroll
        for (int i = 0; i < 4; ++i) {
            afrag[i]      = (__bf16)a0[i];
            afrag[4 + i]  = (__bf16)a1[i];
            afrag[8 + i]  = (__bf16)a2[i];
            afrag[12 + i] = (__bf16)a3[i];
        }

        // ---- 8 WMMAs from current LDS buffer ----
        const __bf16* bb = (const __bf16*)ldsBuf[s & 1];
        #pragma unroll
        for (int t = 0; t < 4; ++t) {
            v16bf brf = *(const v16bf*)(bb + t * 512 + lane * 16);
            accR[t] = __builtin_amdgcn_wmma_f32_16x16x32_bf16(
                false, afrag, false, brf, (short)0, accR[t], false, false);
            v16bf bnf = *(const v16bf*)(bb + 2048 + t * 512 + lane * 16);
            accN[t] = __builtin_amdgcn_wmma_f32_16x16x32_bf16(
                false, afrag, false, bnf, (short)0, accN[t], false, false);
        }
        __syncthreads();          // all reads of buf[s&1] done before reuse
    }

    // ---- epilogue: noisy = logit + noise * softplus(noise_logit) ----
    // C layout: col = lane%16 (+16t), VGPR r -> token row r + 8*laneHalf.
    #pragma unroll
    for (int t = 0; t < 4; ++t) {
        int expert = t * 16 + cl;
        float bR = br[expert];
        float bN = bn[expert];
        #pragma unroll
        for (int r = 0; r < 8; ++r) {
            int token = tokenBase + wave * 16 + r + hi * 8;
            float vR = accR[t][r] + bR;
            float vN = accN[t][r] + bN;
            float sp = (vN > 15.0f) ? vN : log1pf(__expf(vN));
            size_t o = (size_t)expert * N_TOK + token;
            noisy[o] = vR + noise[o] * sp;
        }
    }
}

// ---------------------------------------------------------------------------
// Phase 2: one block (256 threads) per expert.
// Exact top-512 threshold via 4-pass MSB-first radix select on the
// order-preserving uint key, then masked softmax + index emission.
// ---------------------------------------------------------------------------
__device__ __forceinline__ unsigned fkey(unsigned b) {
    return b ^ (((unsigned)((int)b >> 31)) | 0x80000000u);
}

__global__ __launch_bounds__(256) void topk_softmax_kernel(
    const float* __restrict__ noisy,   // [E, N]
    float* __restrict__ router,        // [E, N]
    float* __restrict__ indices)       // [E, K] (written as float values)
{
    __shared__ unsigned hist[256];
    __shared__ float    red[256];
    __shared__ unsigned s_prefix;
    __shared__ unsigned s_rem;
    __shared__ float    s_max, s_denom;
    __shared__ int      s_cnt, s_tie;

    const int e   = blockIdx.x;
    const int tid = threadIdx.x;
    const float* v = noisy + (size_t)e * N_TOK;

    // ---- row max (== top-1) ----
    float m = -INFINITY;
    for (int n = tid; n < N_TOK; n += 256) m = fmaxf(m, v[n]);
    red[tid] = m; __syncthreads();
    for (int s = 128; s > 0; s >>= 1) {
        if (tid < s) red[tid] = fmaxf(red[tid], red[tid + s]);
        __syncthreads();
    }
    if (tid == 0) { s_max = red[0]; s_prefix = 0; s_rem = TOPK; }
    __syncthreads();

    // ---- radix select: find 512th-largest key ----
    for (int pass = 0; pass < 4; ++pass) {
        int shift = 24 - pass * 8;
        unsigned maskHigh = (pass == 0) ? 0u : (0xFFFFFFFFu << (shift + 8));
        hist[tid] = 0;
        __syncthreads();
        unsigned pfx = s_prefix;
        for (int n = tid; n < N_TOK; n += 256) {
            unsigned u = fkey(__float_as_uint(v[n]));
            if ((u & maskHigh) == pfx)
                atomicAdd(&hist[(u >> shift) & 255], 1u);
        }
        __syncthreads();
        if (tid == 0) {
            unsigned cum = 0, remv = s_rem;
            int b = 255;
            for (; b > 0; --b) {
                if (cum + hist[b] >= remv) break;
                cum += hist[b];
            }
            s_rem    = remv - cum;                 // ties to accept at bin b
            s_prefix = pfx | ((unsigned)b << shift);
        }
        __syncthreads();
    }

    const unsigned Tu   = s_prefix;                // threshold key
    const float    maxv = s_max;
    const float    Tval = __uint_as_float(
        (Tu & 0x80000000u) ? (Tu ^ 0x80000000u) : ~Tu);

    // ---- sum of exp over strictly-greater + tie contribution ----
    float ls = 0.0f;
    for (int n = tid; n < N_TOK; n += 256) {
        float val = v[n];
        if (fkey(__float_as_uint(val)) > Tu) ls += __expf(val - maxv);
    }
    red[tid] = ls; __syncthreads();
    for (int s = 128; s > 0; s >>= 1) {
        if (tid < s) red[tid] += red[tid + s];
        __syncthreads();
    }
    if (tid == 0) {
        s_denom = red[0] + (float)s_rem * __expf(Tval - maxv);
        s_cnt = 0;
        s_tie = (int)s_rem;
    }
    __syncthreads();

    const float inv = 1.0f / s_denom;
    float* outIdx = indices + (size_t)e * TOPK;
    float* outRow = router  + (size_t)e * N_TOK;

    for (int n = tid; n < N_TOK; n += 256) {
        float val  = v[n];
        unsigned u = fkey(__float_as_uint(val));
        bool sel = (u > Tu);
        if (!sel && u == Tu)
            sel = (atomicSub(&s_tie, 1) > 0);      // exact tie budget
        float o = 0.0f;
        if (sel) {
            o = __expf(val - maxv) * inv;
            int slot = atomicAdd(&s_cnt, 1);
            if (slot < TOPK) outIdx[slot] = (float)n;
        }
        outRow[n] = o;                              // softmax(-inf) == 0
    }
}

// ---------------------------------------------------------------------------
extern "C" void kernel_launch(void* const* d_in, const int* in_sizes, int n_in,
                              void* d_out, int out_size, void* d_ws, size_t ws_size,
                              hipStream_t stream) {
    const float* x     = (const float*)d_in[0];   // [8,2048,2048] -> [N,D]
    const float* Wr    = (const float*)d_in[1];   // [D,E]
    const float* br    = (const float*)d_in[2];   // [E]
    const float* Wn    = (const float*)d_in[3];   // [D,E]
    const float* bn    = (const float*)d_in[4];   // [E]
    const float* noise = (const float*)d_in[5];   // [E,N]
    // d_in[6] = top_k (== 512, compiled in)

    float* router  = (float*)d_out;                       // [E,N]
    float* indices = router + (size_t)N_EXP * N_TOK;      // [E,K]

    float*  noisy = (float*)d_ws;                                     // 4 MB
    __bf16* frag  = (__bf16*)((char*)d_ws + (size_t)N_EXP * N_TOK * 4); // 512 KB

    pack_w_kernel<<<(KSTEPS * 2048) / 256, 256, 0, stream>>>(Wr, Wn, frag);
    router_gemm_kernel<<<N_TOK / 128, 256, 0, stream>>>(
        x, frag, br, bn, noise, noisy);
    topk_softmax_kernel<<<N_EXP, 256, 0, stream>>>(noisy, router, indices);
}
